// HybridGNNLSTM_42631845380847
// MI455X (gfx1250) — compile-verified, matrix operand order checked by reading, not run on hardware
//
#include <hip/hip_runtime.h>
#include <hip/hip_bf16.h>

typedef __attribute__((ext_vector_type(16))) _Float16 v16h;
typedef __attribute__((ext_vector_type(8)))  float    v8f;
typedef __attribute__((ext_vector_type(4)))  float    v4f;

#define Hh 128
#define Tt 16
#define Ff 32
#define EPSLN 1e-5f

// ---------------------------------------------------------------------------
// Tile loader per CDNA5 WMMA 16-bit A/B layout:
//   lane L (0..31): row = L%16, halves [k0 + (L/16)*8 .. +7] and [+16 .. +23]
// base must point at (row0, k0=0); ld (in halves) must be a multiple of 8.
// ---------------------------------------------------------------------------
__device__ __forceinline__ v16h ld16(const _Float16* base, int ld, int koff) {
  const int lane = threadIdx.x & 31;
  const _Float16* q = base + (size_t)(lane & 15) * ld + koff + ((lane >> 4) << 3);
  v16h v;
  reinterpret_cast<v4f*>(&v)[0] = *reinterpret_cast<const v4f*>(q);
  reinterpret_cast<v4f*>(&v)[1] = *reinterpret_cast<const v4f*>(q + 16);
  return v;
}

__device__ __forceinline__ v8f wmma32(v16h a, v16h b, v8f c) {
  return __builtin_amdgcn_wmma_f32_16x16x32_f16(false, a, false, b, (short)0, c,
                                                false, false);
}

__device__ __forceinline__ v8f zero8() {
  v8f v;
#pragma unroll
  for (int i = 0; i < 8; ++i) v[i] = 0.f;
  return v;
}

__device__ __forceinline__ float sigf(float x) { return 1.f / (1.f + expf(-x)); }

// ---------------------------------------------------------------------------
// Prep kernels
// ---------------------------------------------------------------------------
__global__ void k_vec_add(const float* a, const float* b, float* o, int n) {
  int i = blockIdx.x * blockDim.x + threadIdx.x;
  if (i < n) o[i] = a[i] + b[i];
}
__global__ void k_cvt_f16(const float* in, _Float16* out, int n) {
  int i = blockIdx.x * blockDim.x + threadIdx.x;
  if (i < n) out[i] = (_Float16)in[i];
}
// in: (R,C) row-major fp32 -> out: (C,R) row-major f16 (i.e. transposed)
__global__ void k_transpose_f16(const float* in, _Float16* out, int R, int C) {
  int i = blockIdx.x * blockDim.x + threadIdx.x;
  if (i < R * C) {
    int r = i / C, c = i % C;
    out[(size_t)c * R + r] = (_Float16)in[i];
  }
}

// ---------------------------------------------------------------------------
// LSTM layer 0: x (N,32,16) fp32 -> hs0 (16,N,128) f16.  Weights staged to LDS.
// LDS: Wih[512][40] f16, Whh[512][136] f16, xa[128][40] f16,
//      hb[128][136] f16, c[128][132] f32   => 292,864 B
// Gate GEMMs start from a zero accumulator (inline-0 C operand); bias is
// folded into the scalar elementwise phase to avoid v8f bias splats/spills.
// ---------------------------------------------------------------------------
__global__ void lstm0_kernel(const float* __restrict__ x,
                             const float* __restrict__ Wih,
                             const float* __restrict__ Whh,
                             const float* __restrict__ bsum,
                             _Float16* __restrict__ hs0, int N) {
  extern __shared__ char smem[];
  _Float16* wih = (_Float16*)smem;                 // 512*40
  _Float16* whh = wih + 512 * 40;                  // 512*136
  _Float16* xa  = whh + 512 * 136;                 // 128*40
  _Float16* hb  = xa + 128 * 40;                   // 128*136
  float*    cst = (float*)(hb + 128 * 136);        // 128*132
  const int tid = threadIdx.x, lane = tid & 31, mt = tid >> 5;
  const int n0 = blockIdx.x * 128;

  for (int i = tid; i < 512 * 32; i += 256)
    wih[(i >> 5) * 40 + (i & 31)] = (_Float16)Wih[i];
  for (int i = tid; i < 512 * 128; i += 256)
    whh[(i >> 7) * 136 + (i & 127)] = (_Float16)Whh[i];
  for (int i = tid; i < 128 * 136; i += 256) hb[i] = (_Float16)0.f;
  for (int i = tid; i < 128 * 132; i += 256) cst[i] = 0.f;

  // per-lane scalar bias values (t-invariant): 32 VGPRs, added post-GEMM
  float bias[4][8];
#pragma unroll
  for (int g = 0; g < 4; ++g)
#pragma unroll
    for (int j = 0; j < 8; ++j)
      bias[g][j] = bsum[g * 128 + j * 16 + (lane & 15)];

  __syncthreads();

  const bool wact = (n0 + (mt + 1) * 16) <= N;

  for (int t = 0; t < Tt; ++t) {
    // stage x_t -> LDS (f16), layout (node, feature)
    for (int i = tid; i < 128 * 32; i += 256) {
      int nl = i >> 5, f = i & 31, n = n0 + nl;
      float xv = (n < N) ? x[((size_t)n * Ff + f) * Tt + t] : 0.f;
      xa[nl * 40 + f] = (_Float16)xv;
    }
    __syncthreads();

    v16h Ax, Ah[4];
    if (wact) {
      Ax = ld16(xa + mt * 16 * 40, 40, 0);
#pragma unroll
      for (int kt = 0; kt < 4; ++kt) Ah[kt] = ld16(hb + mt * 16 * 136, 136, kt * 32);
    }
    __syncthreads();  // all A preloads done before hb gets overwritten

    if (wact) {
#pragma unroll
      for (int j = 0; j < 8; ++j) {
        v8f acc[4];
#pragma unroll
        for (int g = 0; g < 4; ++g) {
          const int col0 = g * 128 + j * 16;
          // burst-load all B tiles for this gate, then run the WMMA chain
          v16h B0 = ld16(wih + col0 * 40, 40, 0);
          v16h B1 = ld16(whh + col0 * 136, 136, 0);
          v16h B2 = ld16(whh + col0 * 136, 136, 32);
          v16h B3 = ld16(whh + col0 * 136, 136, 64);
          v16h B4 = ld16(whh + col0 * 136, 136, 96);
          v8f a = zero8();
          a = wmma32(Ax, B0, a);
          a = wmma32(Ah[0], B1, a);
          a = wmma32(Ah[1], B2, a);
          a = wmma32(Ah[2], B3, a);
          a = wmma32(Ah[3], B4, a);
          acc[g] = a;
        }
        const int ch = j * 16 + (lane & 15);
        const int mb = mt * 16 + ((lane >> 4) << 3);
#pragma unroll
        for (int r = 0; r < 8; ++r) {
          int m = mb + r;
          float ii = sigf(acc[0][r] + bias[0][j]);
          float ff = sigf(acc[1][r] + bias[1][j]);
          float gg = tanhf(acc[2][r] + bias[2][j]);
          float oo = sigf(acc[3][r] + bias[3][j]);
          float cn = ff * cst[m * 132 + ch] + ii * gg;
          float hn = oo * tanhf(cn);
          cst[m * 132 + ch] = cn;
          hb[m * 136 + ch] = (_Float16)hn;
          hs0[(size_t)t * N * Hh + (size_t)(n0 + m) * Hh + ch] = (_Float16)hn;
        }
      }
    }
    __syncthreads();
  }
}

// ---------------------------------------------------------------------------
// LSTM layer 1: hs0 (16,N,128) f16 -> fused[:,0:128] f16 (last h).
// Whh in LDS; Wih1 (f16, pre-converted) streamed from global (L2-hot).
// LDS: Whh[512][136] f16, hb[128][136] f16, c[128][132] f32 => 241,664 B
// ---------------------------------------------------------------------------
__global__ void lstm1_kernel(const _Float16* __restrict__ hs0,
                             const _Float16* __restrict__ Wih,  // (512,128) f16
                             const float* __restrict__ Whh,
                             const float* __restrict__ bsum,
                             _Float16* __restrict__ fused, int N) {
  extern __shared__ char smem[];
  _Float16* whh = (_Float16*)smem;                 // 512*136
  _Float16* hb  = whh + 512 * 136;                 // 128*136
  float*    cst = (float*)(hb + 128 * 136);        // 128*132
  const int tid = threadIdx.x, lane = tid & 31, mt = tid >> 5;
  const int n0 = blockIdx.x * 128;

  for (int i = tid; i < 512 * 128; i += 256)
    whh[(i >> 7) * 136 + (i & 127)] = (_Float16)Whh[i];
  for (int i = tid; i < 128 * 136; i += 256) hb[i] = (_Float16)0.f;
  for (int i = tid; i < 128 * 132; i += 256) cst[i] = 0.f;

  float bias[4][8];
#pragma unroll
  for (int g = 0; g < 4; ++g)
#pragma unroll
    for (int j = 0; j < 8; ++j)
      bias[g][j] = bsum[g * 128 + j * 16 + (lane & 15)];

  __syncthreads();

  const bool wact = (n0 + (mt + 1) * 16) <= N;

  for (int t = 0; t < Tt; ++t) {
    v16h Ax[4], Ah[4];
    if (wact) {
      const _Float16* xb = hs0 + (size_t)t * N * Hh + (size_t)(n0 + mt * 16) * Hh;
#pragma unroll
      for (int kt = 0; kt < 4; ++kt) {
        Ax[kt] = ld16(xb, Hh, kt * 32);
        Ah[kt] = ld16(hb + mt * 16 * 136, 136, kt * 32);
      }
    }
    __syncthreads();

    if (wact) {
#pragma unroll
      for (int j = 0; j < 8; ++j) {
        v8f acc[4];
#pragma unroll
        for (int g = 0; g < 4; ++g) {
          const int col0 = g * 128 + j * 16;
          v8f a = zero8();
          // burst 1: 4 global (Wih) B tiles -> 4 WMMAs
          {
            v16h Bx0 = ld16(Wih + (size_t)col0 * Hh, Hh, 0);
            v16h Bx1 = ld16(Wih + (size_t)col0 * Hh, Hh, 32);
            v16h Bx2 = ld16(Wih + (size_t)col0 * Hh, Hh, 64);
            v16h Bx3 = ld16(Wih + (size_t)col0 * Hh, Hh, 96);
            a = wmma32(Ax[0], Bx0, a);
            a = wmma32(Ax[1], Bx1, a);
            a = wmma32(Ax[2], Bx2, a);
            a = wmma32(Ax[3], Bx3, a);
          }
          // burst 2: 4 LDS (Whh) B tiles -> 4 WMMAs
          {
            v16h Bh0 = ld16(whh + col0 * 136, 136, 0);
            v16h Bh1 = ld16(whh + col0 * 136, 136, 32);
            v16h Bh2 = ld16(whh + col0 * 136, 136, 64);
            v16h Bh3 = ld16(whh + col0 * 136, 136, 96);
            a = wmma32(Ah[0], Bh0, a);
            a = wmma32(Ah[1], Bh1, a);
            a = wmma32(Ah[2], Bh2, a);
            a = wmma32(Ah[3], Bh3, a);
          }
          acc[g] = a;
        }
        const int ch = j * 16 + (lane & 15);
        const int mb = mt * 16 + ((lane >> 4) << 3);
#pragma unroll
        for (int r = 0; r < 8; ++r) {
          int m = mb + r;
          float ii = sigf(acc[0][r] + bias[0][j]);
          float ff = sigf(acc[1][r] + bias[1][j]);
          float gg = tanhf(acc[2][r] + bias[2][j]);
          float oo = sigf(acc[3][r] + bias[3][j]);
          float cn = ff * cst[m * 132 + ch] + ii * gg;
          float hn = oo * tanhf(cn);
          cst[m * 132 + ch] = cn;
          hb[m * 136 + ch] = (_Float16)hn;
          if (t == Tt - 1)
            fused[(size_t)(n0 + m) * 256 + ch] = (_Float16)hn;
        }
      }
    }
    __syncthreads();
  }
}

// ---------------------------------------------------------------------------
// Generic WMMA GEMM (K compile-time): C(M,Nc) = relu?(A(M,K)f16 @ Bt(Nc,K)^T + b)
// Block = 128 rows (8 waves x 16).  Nc <= 128, K in {128, 256}.
// A tiles register-resident across the N loop; B tiles burst-loaded.
// ---------------------------------------------------------------------------
template <int K>
__global__ void wmma_gemm_nt(const _Float16* __restrict__ A, int lda,
                             const _Float16* __restrict__ Bt, int ldb,
                             const float* __restrict__ bias,
                             float* __restrict__ Cf,
                             _Float16* __restrict__ Ch, int ldch,
                             int M, int Nc, int relu) {
  constexpr int KT = K / 32;
  const int m0 = blockIdx.x * 128 + (threadIdx.x >> 5) * 16;
  if (m0 + 16 > M) return;  // wave-uniform
  const int lane = threadIdx.x & 31;
  const int ntiles = Nc >> 4;

  v16h At[KT];
#pragma unroll
  for (int kt = 0; kt < KT; ++kt) At[kt] = ld16(A + (size_t)m0 * lda, lda, kt * 32);

  for (int nt = 0; nt < ntiles; ++nt) {
    v16h Bm[KT];
#pragma unroll
    for (int kt = 0; kt < KT; ++kt)
      Bm[kt] = ld16(Bt + (size_t)(nt * 16) * ldb, ldb, kt * 32);
    v8f acc = zero8();
#pragma unroll
    for (int kt = 0; kt < KT; ++kt) acc = wmma32(At[kt], Bm[kt], acc);

    float b = bias ? bias[nt * 16 + (lane & 15)] : 0.f;
    const int n = nt * 16 + (lane & 15);
    const int mb = m0 + ((lane >> 4) << 3);
#pragma unroll
    for (int r = 0; r < 8; ++r) {
      float v = acc[r] + b;
      if (relu) v = fmaxf(v, 0.f);
      if (Cf) Cf[(size_t)(mb + r) * Nc + n] = v;
      if (Ch) Ch[(size_t)(mb + r) * ldch + n] = (_Float16)v;
    }
  }
}

// ---------------------------------------------------------------------------
// GCN helpers
// ---------------------------------------------------------------------------
__global__ void k_set1(float* p, int n) {
  int i = blockIdx.x * blockDim.x + threadIdx.x;
  if (i < n) p[i] = 1.0f;
}
__global__ void k_deg_edge(const int* __restrict__ ei, const float* __restrict__ w,
                           float* __restrict__ deg, int E) {
  int e = blockIdx.x * blockDim.x + threadIdx.x;
  if (e < E) atomicAdd(&deg[ei[E + e]], w[e]);
}
__global__ void k_rsqrt(float* p, int n) {
  int i = blockIdx.x * blockDim.x + threadIdx.x;
  if (i < n) { float v = p[i]; p[i] = (v > 0.f) ? rsqrtf(v) : 0.f; }
}
// agg[n][c] = bias[c] + dinv[n]^2 * h[n][c]   (self-loop, w=1)
__global__ void k_gcn_init(float* __restrict__ agg, const float* __restrict__ bias,
                           const float* __restrict__ dinv,
                           const float* __restrict__ h, int N) {
  int i = blockIdx.x * blockDim.x + threadIdx.x;
  if (i < N * Hh) {
    int n = i >> 7, c = i & 127;
    float di = dinv[n];
    agg[i] = bias[c] + di * di * h[i];
  }
}
// one wave per edge; 4 channels per lane
__global__ void k_gcn_edge(const int* __restrict__ ei, const float* __restrict__ w,
                           const float* __restrict__ dinv,
                           const float* __restrict__ h, float* __restrict__ agg,
                           int E) {
  int gid = blockIdx.x * blockDim.x + threadIdx.x;
  int e = gid >> 5;
  if (e >= E) return;
  int lane = gid & 31;
  int s = ei[e], d = ei[E + e];
  float norm = dinv[s] * w[e] * dinv[d];
#pragma unroll
  for (int i = 0; i < 4; ++i) {
    int c = lane + i * 32;
    atomicAdd(&agg[(size_t)d * Hh + c], h[(size_t)s * Hh + c] * norm);
  }
}
// LayerNorm over H=128 (one wave per node), optional ReLU, f16 output
__global__ void k_ln_f16(const float* __restrict__ X, const float* __restrict__ g,
                         const float* __restrict__ be, _Float16* __restrict__ Y,
                         int ldy, int relu, int N) {
  int n = blockIdx.x * 8 + (threadIdx.x >> 5);
  if (n >= N) return;
  int lane = threadIdx.x & 31;
  float v[4], s = 0.f;
#pragma unroll
  for (int i = 0; i < 4; ++i) { v[i] = X[(size_t)n * Hh + i * 32 + lane]; s += v[i]; }
#pragma unroll
  for (int m = 16; m > 0; m >>= 1) s += __shfl_xor(s, m, 32);
  float mu = s * (1.f / Hh), var = 0.f;
#pragma unroll
  for (int i = 0; i < 4; ++i) { float d = v[i] - mu; var += d * d; }
#pragma unroll
  for (int m = 16; m > 0; m >>= 1) var += __shfl_xor(var, m, 32);
  float inv = rsqrtf(var * (1.f / Hh) + EPSLN);
#pragma unroll
  for (int i = 0; i < 4; ++i) {
    int c = i * 32 + lane;
    float y = (v[i] - mu) * inv * g[c] + be[c];
    if (relu) y = fmaxf(y, 0.f);
    Y[(size_t)n * ldy + c] = (_Float16)y;
  }
}
// out[n] = dot(hid[n][0:64], Wp2) + bp2  (one wave per node)
__global__ void k_final(const _Float16* __restrict__ hid, const float* __restrict__ w,
                        const float* __restrict__ b, float* __restrict__ out, int N) {
  int n = blockIdx.x * 8 + (threadIdx.x >> 5);
  if (n >= N) return;
  int lane = threadIdx.x & 31;
  float s = (float)hid[(size_t)n * 64 + lane] * w[lane] +
            (float)hid[(size_t)n * 64 + 32 + lane] * w[32 + lane];
#pragma unroll
  for (int m = 16; m > 0; m >>= 1) s += __shfl_xor(s, m, 32);
  if (lane == 0) out[n] = s + b[0];
}

// ---------------------------------------------------------------------------
extern "C" void kernel_launch(void* const* d_in, const int* in_sizes, int n_in,
                              void* d_out, int out_size, void* d_ws, size_t ws_size,
                              hipStream_t stream) {
  const float* x    = (const float*)d_in[0];
  const int*   ei   = (const int*)d_in[1];
  const float* ew   = (const float*)d_in[2];
  const float* Wih0 = (const float*)d_in[3];
  const float* Whh0 = (const float*)d_in[4];
  const float* bih0 = (const float*)d_in[5];
  const float* bhh0 = (const float*)d_in[6];
  const float* Wih1 = (const float*)d_in[7];
  const float* Whh1 = (const float*)d_in[8];
  const float* bih1 = (const float*)d_in[9];
  const float* bhh1 = (const float*)d_in[10];
  const float* Wg1  = (const float*)d_in[11];
  const float* bg1  = (const float*)d_in[12];
  const float* g1   = (const float*)d_in[13];
  const float* be1  = (const float*)d_in[14];
  const float* Wg2  = (const float*)d_in[15];
  const float* bg2  = (const float*)d_in[16];
  const float* g2   = (const float*)d_in[17];
  const float* be2  = (const float*)d_in[18];
  const float* Wp1  = (const float*)d_in[19];
  const float* bp1  = (const float*)d_in[20];
  const float* Wp2  = (const float*)d_in[21];
  const float* bp2  = (const float*)d_in[22];
  float* out = (float*)d_out;

  const int N = in_sizes[0] / (Ff * Tt);   // 50000
  const int E = in_sizes[2];               // 800000

  // workspace carving
  char* ws = (char*)d_ws;
  size_t off = 0;
  auto carve = [&](size_t bytes) -> void* {
    void* p = ws + off;
    off += (bytes + 255) & ~(size_t)255;
    return p;
  };
  _Float16* hs0   = (_Float16*)carve((size_t)Tt * N * Hh * 2);
  _Float16* fused = (_Float16*)carve((size_t)N * 256 * 2);
  _Float16* hln1  = (_Float16*)carve((size_t)N * Hh * 2);
  _Float16* hid   = (_Float16*)carve((size_t)N * 64 * 2);
  _Float16* Wih1h = (_Float16*)carve((size_t)512 * 128 * 2);
  _Float16* Wg1T  = (_Float16*)carve((size_t)128 * 128 * 2);
  _Float16* Wg2T  = (_Float16*)carve((size_t)128 * 128 * 2);
  _Float16* Wp1T  = (_Float16*)carve((size_t)64 * 256 * 2);
  float* b0sum = (float*)carve(512 * 4);
  float* b1sum = (float*)carve(512 * 4);
  float* dinv  = (float*)carve((size_t)N * 4);
  float* hmat  = (float*)carve((size_t)N * Hh * 4);
  float* agg   = (float*)carve((size_t)N * Hh * 4);

  const int nb = (N + 127) / 128;  // 391

  // ---- prep ----
  k_vec_add<<<2, 256, 0, stream>>>(bih0, bhh0, b0sum, 512);
  k_vec_add<<<2, 256, 0, stream>>>(bih1, bhh1, b1sum, 512);
  k_cvt_f16<<<256, 256, 0, stream>>>(Wih1, Wih1h, 512 * 128);
  k_transpose_f16<<<64, 256, 0, stream>>>(Wg1, Wg1T, 128, 128);
  k_transpose_f16<<<64, 256, 0, stream>>>(Wg2, Wg2T, 128, 128);
  k_transpose_f16<<<64, 256, 0, stream>>>(Wp1, Wp1T, 256, 64);

  // ---- temporal encoder (WMMA LSTM, 2 layers) ----
  const size_t sh0 = (512 * 40 + 512 * 136 + 128 * 40 + 128 * 136) * 2 + 128 * 132 * 4;
  const size_t sh1 = (512 * 136 + 128 * 136) * 2 + 128 * 132 * 4;
  lstm0_kernel<<<nb, 256, sh0, stream>>>(x, Wih0, Whh0, b0sum, hs0, N);
  lstm1_kernel<<<nb, 256, sh1, stream>>>(hs0, Wih1h, Whh1, b1sum, fused, N);

  // ---- degrees (with self-loops) ----
  k_set1<<<(N + 255) / 256, 256, 0, stream>>>(dinv, N);
  k_deg_edge<<<(E + 255) / 256, 256, 0, stream>>>(ei, ew, dinv, E);
  k_rsqrt<<<(N + 255) / 256, 256, 0, stream>>>(dinv, N);

  // ---- GCN layer 1 ----
  wmma_gemm_nt<128><<<nb, 256, 0, stream>>>(fused, 256, Wg1T, 128, nullptr,
                                            hmat, nullptr, 0, N, 128, 0);
  k_gcn_init<<<(N * Hh + 255) / 256, 256, 0, stream>>>(agg, bg1, dinv, hmat, N);
  k_gcn_edge<<<((size_t)E * 32 + 255) / 256, 256, 0, stream>>>(ei, ew, dinv, hmat, agg, E);
  k_ln_f16<<<(N + 7) / 8, 256, 0, stream>>>(agg, g1, be1, hln1, 128, 1, N);

  // ---- GCN layer 2 ----
  wmma_gemm_nt<128><<<nb, 256, 0, stream>>>(hln1, 128, Wg2T, 128, nullptr,
                                            hmat, nullptr, 0, N, 128, 0);
  k_gcn_init<<<(N * Hh + 255) / 256, 256, 0, stream>>>(agg, bg2, dinv, hmat, N);
  k_gcn_edge<<<((size_t)E * 32 + 255) / 256, 256, 0, stream>>>(ei, ew, dinv, hmat, agg, E);
  k_ln_f16<<<(N + 7) / 8, 256, 0, stream>>>(agg, g2, be2, fused + 128, 256, 0, N);

  // ---- predictor ----
  wmma_gemm_nt<256><<<nb, 256, 0, stream>>>(fused, 256, Wp1T, 256, bp1,
                                            nullptr, hid, 64, N, 64, 1);
  k_final<<<(N + 7) / 8, 256, 0, stream>>>(hid, Wp2, bp2, out, N);

  (void)n_in; (void)out_size; (void)ws_size;
}
